// TemporalLayer_39685497815924
// MI455X (gfx1250) — matrix operand order repaired and checked by reading, not compiled, not run on hardware
//
#include <hip/hip_runtime.h>
#include <stdint.h>
#include <stddef.h>

// ---------------- problem constants ----------------
#define BATCH 32
#define SEQL  512
#define DIM   1024
#define E3    3072            // 3*DIM
#define KT    32              // K per bf16 WMMA
#define NKT   (DIM / KT)      // 32 k-tiles over DIM
#define NT3   (E3 / 16)       // 192 column tiles over 3*DIM
#define ROWS  (BATCH * SEQL)  // 16384 rows of x
#define NBLK  64              // persistent scan blocks (one per dt)
#define FRAG_DW (32 * 8)      // dwords per B fragment (32 lanes x 8)
#define SMEM_DW (3 * NKT * FRAG_DW)  // 24576 dwords = 96 KB per block

typedef __attribute__((ext_vector_type(16))) __bf16 v16bf;
typedef __attribute__((ext_vector_type(8)))  float  v8f;

// ---------------- bf16 helpers (round-to-nearest-even) ----------------
__device__ __forceinline__ uint32_t f32_to_bf16_bits(float f) {
  union { float f; uint32_t u; } c; c.f = f;
  return (c.u + 0x7FFFu + ((c.u >> 16) & 1u)) >> 16;
}
__device__ __forceinline__ uint32_t pack2_bf16(float lo, float hi) {
  return (f32_to_bf16_bits(hi) << 16) | (f32_to_bf16_bits(lo) & 0xFFFFu);
}

__device__ __forceinline__ v8f wmma_bf16(v16bf a, v16bf b, v8f c) {
  return __builtin_amdgcn_wmma_f32_16x16x32_bf16(
      /*neg_a=*/false, a, /*neg_b=*/false, b,
      /*c_mod=*/(short)0, c, /*reuse_a=*/false, /*reuse_b=*/false);
}

// ---------------- fragment loaders ----------------
// A-matrix 16x32 bf16 (ISA 7.12.2): lane m = lane&15;
//   v in 0..3 : K = 2v        + (lane>=16 ? 8 : 0)
//   v in 4..7 : K = 16+2(v-4) + (lane>=16 ? 8 : 0)
// Source: row-major bf16 viewed as packed uint32 pairs (even K aligned)
// -> per lane two contiguous 4-dword runs => 2x global_load_b128.
__device__ __forceinline__ v16bf load_a_frag(const uint32_t* __restrict__ base_pairs,
                                             int row_stride_pairs,
                                             int rowbase, int kbase, int lane) {
  union { v16bf v; uint32_t u[8]; } a;
  const int m = rowbase + (lane & 15);
  const int khalf = (lane >> 4) << 3; // 0 or 8
  const uint32_t* rp = base_pairs + (size_t)m * row_stride_pairs;
#pragma unroll
  for (int v = 0; v < 8; ++v) {
    const int k0 = kbase + ((v < 4) ? (2 * v) : (16 + 2 * (v - 4))) + khalf;
    a.u[v] = rp[k0 >> 1];
  }
  return a.v;
}

// Pre-packed B fragment in global: 8 consecutive dwords per lane.
__device__ __forceinline__ v16bf load_b_frag(const uint32_t* __restrict__ frags,
                                             int nt, int kt, int lane) {
  union { v16bf v; uint32_t u[8]; } b;
  const uint32_t* p = frags + ((((size_t)nt * NKT) + kt) * 32 + lane) * 8;
#pragma unroll
  for (int v = 0; v < 8; ++v) b.u[v] = p[v];
  return b.v;
}

// B fragment from LDS (gate-major): 2x ds_load_b128 per lane.
__device__ __forceinline__ v16bf load_b_lds(const uint32_t* sm, int g, int kt, int lane) {
  union { v16bf v; uint32_t u[8]; } b;
  const uint32_t* p = sm + ((g * NKT + kt) * 32 + lane) * 8;
#pragma unroll
  for (int v = 0; v < 8; ++v) b.u[v] = p[v];
  return b.v;
}

// ---------------- setup kernels ----------------
// Pack w (E3 x DIM fp32 row-major; B[k][n] = w[n][k]) into B-fragment order.
__global__ void pack_w_frags(const float* __restrict__ w, uint32_t* __restrict__ out) {
  const int idx = blockIdx.x * blockDim.x + threadIdx.x;
  const int total = NT3 * NKT * FRAG_DW;
  if (idx >= total) return;
  const int v    = idx & 7;
  const int lane = (idx >> 3) & 31;
  const int kt   = (idx >> 8) % NKT;
  const int nt   = (idx >> 8) / NKT;
  const int n  = nt * 16 + (lane & 15);
  const int k0 = kt * 32 + ((lane >= 16) ? 8 : 0) + ((v < 4) ? (2 * v) : (16 + 2 * (v - 4)));
  const float* wr = w + (size_t)n * DIM + k0;
  out[idx] = pack2_bf16(wr[0], wr[1]);
}

// Convert x (fp32) -> packed bf16 pairs, row-major (ROWS x DIM).
__global__ void cvt_x_bf16(const float* __restrict__ x, uint32_t* __restrict__ xb) {
  const int i = blockIdx.x * blockDim.x + threadIdx.x;
  const int npairs = ROWS * DIM / 2;
  if (i >= npairs) return;
  xb[i] = pack2_bf16(x[2 * i], x[2 * i + 1]);
}

// Initialize h state: fp32 copy + bf16 copy of prev_state.
__global__ void init_h(const float* __restrict__ h0, float* __restrict__ hf,
                       uint16_t* __restrict__ hb) {
  const int i = blockIdx.x * blockDim.x + threadIdx.x;
  if (i >= BATCH * DIM) return;
  const float v = h0[i];
  hf[i] = v;
  hb[i] = (uint16_t)f32_to_bf16_bits(v);
}

// ---------------- phase 1: xg = x @ w_ih^T + b_ih, stored (L, B, 3D) ----------------
__global__ void __launch_bounds__(128)
gemm_xg(const uint32_t* __restrict__ xb, const uint32_t* __restrict__ wihF,
        const float* __restrict__ b_ih, float* __restrict__ xg) {
  const int wave = blockIdx.x * (blockDim.x >> 5) + (threadIdx.x >> 5);
  const int lane = threadIdx.x & 31;
  const int rt = wave / NT3;   // 0..1023 row tile over (B*L)
  const int nt = wave % NT3;   // 0..191  col tile
  v8f acc = {};
  for (int kt = 0; kt < NKT; ++kt) {
    v16bf a = load_a_frag(xb, DIM / 2, rt * 16, kt * KT, lane);
    v16bf b = load_b_frag(wihF, nt, kt, lane);
    acc = wmma_bf16(a, b, acc);
  }
  const int col  = nt * 16 + (lane & 15);
  const float bias = b_ih[col];
  const int mbase = rt * 16 + ((lane >= 16) ? 8 : 0);
#pragma unroll
  for (int i = 0; i < 8; ++i) {
    const int m = mbase + i;            // row in (B*L) order: m = b*SEQL + t
    const int bb = m / SEQL;
    const int tt = m % SEQL;
    xg[((size_t)tt * BATCH + bb) * E3 + col] = acc[i] + bias;  // (L, B, 3D)
  }
}

// ---------------- phase 2: persistent GRU scan ----------------
// 64 blocks x 64 threads (2 waves). Block owns column tile dt = blockIdx.x;
// wave 0 -> batch rows 0..15, wave 1 -> rows 16..31. The 96 KB of w_hh
// B-fragments for (dt, dt+64, dt+128) are loaded into LDS once and reused
// for all 512 steps. Device-wide barrier between steps (monotonic counter).
__global__ void __launch_bounds__(64)
gru_scan(uint16_t* __restrict__ hbA, float* __restrict__ hfA,
         uint16_t* __restrict__ hbB, float* __restrict__ hfB,
         const uint32_t* __restrict__ whhF, const float* __restrict__ b_hh,
         const float* __restrict__ xg, float* __restrict__ ys,
         unsigned* __restrict__ counter) {
  extern __shared__ uint32_t smem[];          // SMEM_DW dwords = 96 KB
  const int dt   = blockIdx.x;                // 0..63
  const int lane = threadIdx.x & 31;
  const int rt   = threadIdx.x >> 5;          // wave id = row tile (0..1)

  // ---- one-time preload of this block's w_hh fragments into LDS ----
  for (int idx = threadIdx.x; idx < SMEM_DW; idx += blockDim.x) {
    const int g  = idx / (NKT * FRAG_DW);            // gate 0..2
    const int r  = idx % (NKT * FRAG_DW);
    const int kt = r / FRAG_DW;
    const int o  = r % FRAG_DW;                      // lane*8 + v
    const int nt = dt + g * (DIM / 16);
    smem[idx] = whhF[(((size_t)nt * NKT) + kt) * FRAG_DW + o];
  }
  __syncthreads();

  const int j = dt * 16 + (lane & 15);
  const float bhr = b_hh[j];
  const float bhz = b_hh[DIM + j];
  const float bhn = b_hh[2 * DIM + j];
  const int mbase = rt * 16 + ((lane >= 16) ? 8 : 0);

  for (int t = 0; t < SEQL; ++t) {
    const uint16_t* hb_in  = (t & 1) ? hbB : hbA;
    const float*    hf_in  = (t & 1) ? hfB : hfA;
    uint16_t*       hb_out = (t & 1) ? hbA : hbB;
    float*          hf_out = (t & 1) ? hfA : hfB;

    const uint32_t* hpairs = (const uint32_t*)hb_in;
    v8f cr = {}, cz = {}, cn = {};
    for (int kt = 0; kt < NKT; ++kt) {
      v16bf a  = load_a_frag(hpairs, DIM / 2, rt * 16, kt * KT, lane);
      v16bf br = load_b_lds(smem, 0, kt, lane);
      v16bf bz = load_b_lds(smem, 1, kt, lane);
      v16bf bn = load_b_lds(smem, 2, kt, lane);
      cr = wmma_bf16(a, br, cr);
      cz = wmma_bf16(a, bz, cz);
      cn = wmma_bf16(a, bn, cn);
    }

    const size_t xstep = (size_t)t * BATCH * E3;   // (L, B, 3D) layout
#pragma unroll
    for (int i = 0; i < 8; ++i) {
      const int b = mbase + i;
      const size_t xrow = xstep + (size_t)b * E3;
      const float xr = xg[xrow + j];
      const float xz = xg[xrow + DIM + j];
      const float xn = xg[xrow + 2 * DIM + j];
      const float hold = hf_in[b * DIM + j];
      const float r = 1.0f / (1.0f + __expf(-(xr + cr[i] + bhr)));
      const float z = 1.0f / (1.0f + __expf(-(xz + cz[i] + bhz)));
      const float n = tanhf(xn + r * (cn[i] + bhn));
      const float hv = (1.0f - z) * n + z * hold;
      hf_out[b * DIM + j] = hv;
      hb_out[b * DIM + j] = (uint16_t)f32_to_bf16_bits(hv);
      ys[((size_t)b * SEQL + t) * DIM + j] = hv;
      // stream-prefetch next step's gate pre-activations
      if (t + 1 < SEQL) {
        const float* nx = xg + xrow + (size_t)BATCH * E3;
        __builtin_prefetch(nx + j, 0, 1);
        __builtin_prefetch(nx + DIM + j, 0, 1);
        __builtin_prefetch(nx + 2 * DIM + j, 0, 1);
      }
    }

    // ---- device-wide step barrier (monotonic counter, 64 arrivals/step) ----
    __threadfence();                 // make this thread's h stores visible (agent)
    __syncthreads();                 // all threads in block done before arriving
    if (threadIdx.x == 0) {
      __hip_atomic_fetch_add(counter, 1u, __ATOMIC_RELEASE, __HIP_MEMORY_SCOPE_AGENT);
      const unsigned target = (unsigned)(t + 1) * NBLK;
      while (__hip_atomic_load(counter, __ATOMIC_ACQUIRE, __HIP_MEMORY_SCOPE_AGENT) < target)
        __builtin_amdgcn_s_sleep(1);
    }
    __syncthreads();                 // release the block once step t is globally done
  }
}

// ---------------- launcher ----------------
extern "C" void kernel_launch(void* const* d_in, const int* in_sizes, int n_in,
                              void* d_out, int out_size, void* d_ws, size_t ws_size,
                              hipStream_t stream) {
  const float* x    = (const float*)d_in[0];
  const float* h0   = (const float*)d_in[1];
  const float* w_ih = (const float*)d_in[2];
  const float* w_hh = (const float*)d_in[3];
  const float* b_ih = (const float*)d_in[4];
  const float* b_hh = (const float*)d_in[5];

  float* ys     = (float*)d_out;                       // (B, L, D)
  float* h_last = ys + (size_t)BATCH * SEQL * DIM;     // (B, D)

  // workspace carve-up
  char* p = (char*)d_ws;
  float*    xg   = (float*)p;    p += (size_t)ROWS * E3 * sizeof(float);           // 192 MB
  uint32_t* xb   = (uint32_t*)p; p += (size_t)ROWS * (DIM / 2) * sizeof(uint32_t); // 32 MB
  uint32_t* wihF = (uint32_t*)p; p += (size_t)NT3 * NKT * FRAG_DW * sizeof(uint32_t); // 6 MB
  uint32_t* whhF = (uint32_t*)p; p += (size_t)NT3 * NKT * FRAG_DW * sizeof(uint32_t); // 6 MB
  float*    hf0  = (float*)p;    p += (size_t)BATCH * DIM * sizeof(float);
  float*    hf1  = (float*)p;    p += (size_t)BATCH * DIM * sizeof(float);
  uint16_t* hb0  = (uint16_t*)p; p += (size_t)BATCH * DIM * sizeof(uint16_t);
  uint16_t* hb1  = (uint16_t*)p; p += (size_t)BATCH * DIM * sizeof(uint16_t);
  unsigned* ctr  = (unsigned*)p; p += 256;

  // reset the scan barrier counter (deterministic per call)
  (void)hipMemsetAsync(ctr, 0, sizeof(unsigned), stream);

  // setup: weight fragment packing, x conversion, h init (all parallel)
  {
    const int total = NT3 * NKT * FRAG_DW;
    pack_w_frags<<<(total + 255) / 256, 256, 0, stream>>>(w_ih, wihF);
    pack_w_frags<<<(total + 255) / 256, 256, 0, stream>>>(w_hh, whhF);
  }
  {
    const int npairs = ROWS * DIM / 2;
    cvt_x_bf16<<<(npairs + 255) / 256, 256, 0, stream>>>(x, xb);
  }
  init_h<<<(BATCH * DIM + 255) / 256, 256, 0, stream>>>(h0, hf0, hb0);

  // phase 1: big WMMA GEMM for all input gates
  {
    const int tiles = (ROWS / 16) * NT3;     // 196608 waves
    gemm_xg<<<tiles / 4, 128, 0, stream>>>(xb, wihF, b_ih, xg);
  }

  // phase 2: single persistent kernel runs all 512 steps
  gru_scan<<<NBLK, 64, SMEM_DW * sizeof(uint32_t), stream>>>(
      hb0, hf0, hb1, hf1, whhF, b_hh, xg, ys, ctr);

  // after 512 steps the final state is back in slot 0
  (void)hipMemcpyAsync(h_last, hf0, (size_t)BATCH * DIM * sizeof(float),
                       hipMemcpyDeviceToDevice, stream);
}
